// Attention_17927193494198
// MI455X (gfx1250) — matrix operand order repaired and checked by reading, not compiled, not run on hardware
//
#include <hip/hip_runtime.h>
#include <math.h>

// ---------------------------------------------------------------------------
// Types for CDNA5 WMMA (wave32): v16bf A/B fragments, v8f C/D fragments.
// ---------------------------------------------------------------------------
typedef unsigned short u16;
typedef __attribute__((ext_vector_type(16))) __bf16 v16bf;
typedef __attribute__((ext_vector_type(8)))  __bf16 v8bf;
typedef __attribute__((ext_vector_type(8)))  float  v8f;

__device__ __forceinline__ u16 f2bf(float f) {
  unsigned u = __float_as_uint(f);
  u += 0x7FFFu + ((u >> 16) & 1u);   // round-to-nearest-even
  return (u16)(u >> 16);
}

// Build a 16-element bf16 fragment from two contiguous 8-element (16B) chunks.
__device__ __forceinline__ v16bf load_frag(const u16* lo, const u16* hi) {
  union { v16bf v; v8bf h[2]; } u;
  u.h[0] = *(const v8bf*)lo;
  u.h[1] = *(const v8bf*)hi;
  return u.v;
}

__device__ __forceinline__ v8f wmma_bf16(v16bf a, v16bf b, v8f c) {
  return __builtin_amdgcn_wmma_f32_16x16x32_bf16(
      /*neg_a=*/false, a, /*neg_b=*/false, b,
      /*c_mod=*/(short)0, c, /*reuse_a=*/false, /*reuse_b=*/false);
}

// CDNA5 async global->LDS copy, 16 bytes per lane, tracked by ASYNCcnt.
// VDST VGPR carries the wave-relative LDS byte address (= low 32 bits of the
// generic __shared__ pointer, per the LDS aperture rule ADDR[31:0]).
__device__ __forceinline__ void async_copy_b128(u16* lds_dst, const u16* gsrc) {
  unsigned lds_off = (unsigned)(size_t)lds_dst;
  unsigned long long ga = (unsigned long long)(size_t)gsrc;
  asm volatile("global_load_async_to_lds_b128 %0, %1, off"
               :: "v"(lds_off), "v"(ga) : "memory");
}
__device__ __forceinline__ void wait_async_zero() {
  asm volatile("s_wait_asynccnt 0" ::: "memory");
}

// ---------------------------------------------------------------------------
// LayerNorm over D=1024, one row per block (256 threads), bf16 output.
// ---------------------------------------------------------------------------
__global__ __launch_bounds__(256) void ln_kernel(
    const float* __restrict__ x, const float* __restrict__ g,
    const float* __restrict__ beta, u16* __restrict__ out) {
  const int D = 1024;
  int row = blockIdx.x;
  const float* xr = x + (size_t)row * D;
  float v[4], s = 0.f, s2 = 0.f;
  for (int i = 0; i < 4; ++i) {
    v[i] = xr[threadIdx.x + i * 256];
    s += v[i]; s2 += v[i] * v[i];
  }
  __shared__ float red[256], red2[256];
  red[threadIdx.x] = s; red2[threadIdx.x] = s2;
  __syncthreads();
  for (int st = 128; st > 0; st >>= 1) {
    if ((int)threadIdx.x < st) {
      red[threadIdx.x]  += red[threadIdx.x + st];
      red2[threadIdx.x] += red2[threadIdx.x + st];
    }
    __syncthreads();
  }
  float mean = red[0] * (1.f / 1024.f);
  float var  = red2[0] * (1.f / 1024.f) - mean * mean;
  float inv  = rsqrtf(var + 1e-5f);
  u16* o = out + (size_t)row * D;
  for (int i = 0; i < 4; ++i) {
    int c = threadIdx.x + i * 256;
    o[c] = f2bf((v[i] - mean) * inv * g[c] + beta[c]);
  }
}

// ---------------------------------------------------------------------------
// Transpose + fp32->bf16 convert: W[K,N] fp32  ->  Wt[N,K] bf16.
// block (32,8), grid (N/32, K/32). LDS-tiled for coalescing both sides.
// ---------------------------------------------------------------------------
__global__ void trans_cvt_kernel(const float* __restrict__ W,
                                 u16* __restrict__ Wt, int K, int N) {
  __shared__ float tile[32][33];
  int tx = threadIdx.x, ty = threadIdx.y;
  int n0 = blockIdx.x * 32, k0 = blockIdx.y * 32;
  for (int i = 0; i < 4; ++i)
    tile[ty + i * 8][tx] = W[(size_t)(k0 + ty + i * 8) * N + (n0 + tx)];
  __syncthreads();
  for (int i = 0; i < 4; ++i)
    Wt[(size_t)(n0 + ty + i * 8) * K + (k0 + tx)] = f2bf(tile[tx][ty + i * 8]);
}

// ---------------------------------------------------------------------------
// Per-head V transpose: qkv[b,p, 2D + h*64 + d] -> vt[(b*H+h)*64 + d, p]
// so the PV WMMA B-fragment (K = key index) is contiguous in memory.
// block (32,8), grid (P/32, 64/32, B*H).
// ---------------------------------------------------------------------------
__global__ void vtrans_kernel(const u16* __restrict__ qkv, u16* __restrict__ vt) {
  __shared__ u16 tile[32][33];
  const int P = 1024;
  int tx = threadIdx.x, ty = threadIdx.y;
  int bh = blockIdx.z; int b = bh >> 4; int h = bh & 15;
  int p0 = blockIdx.x * 32, d0 = blockIdx.y * 32;
  for (int i = 0; i < 4; ++i) {
    int p = p0 + ty + i * 8;
    tile[ty + i * 8][tx] =
        qkv[(size_t)(b * P + p) * 3072 + 2048 + h * 64 + d0 + tx];
  }
  __syncthreads();
  for (int i = 0; i < 4; ++i) {
    int d = d0 + ty + i * 8;
    vt[(size_t)(bh * 64 + d) * P + p0 + tx] = tile[tx][ty + i * 8];
  }
}

// ---------------------------------------------------------------------------
// Tiled bf16 WMMA GEMM:  out = act(A[M,K] * Wt[N,K]^T + bias) (+resid)
//   A row-major bf16, Wt is W transposed ([N,K]) bf16.
//   8 waves/block: 4(M) x 2(N); each wave owns a 32x64 tile = 8 C fragments.
//   Software-pipelined: fragments for chunk k+32 are issued before the WMMAs
//   of chunk k, so s_wait_loadcnt only ever waits on the older clause while
//   the younger loads stay in flight (LOADcnt decrements in issue order).
//   act: 0 = none, 1 = exact GELU. outF fp32 and/or outB bf16.
// ---------------------------------------------------------------------------
__global__ __launch_bounds__(256) void gemm_kernel(
    const u16* __restrict__ A, const u16* __restrict__ Wt,
    const float* __restrict__ bias, const float* __restrict__ resid,
    float* __restrict__ outF, u16* __restrict__ outB,
    int M, int N, int K, int act) {
  int lane = threadIdx.x & 31;
  int w    = threadIdx.x >> 5;
  int lr   = lane & 15, half = lane >> 4;
  int m_base = blockIdx.y * 128 + (w & 3) * 32;
  int n_base = blockIdx.x * 128 + (w >> 2) * 64;
  (void)M;

  v8f c[2][4];
  for (int mi = 0; mi < 2; ++mi)
    for (int ni = 0; ni < 4; ++ni)
      for (int i = 0; i < 8; ++i) c[mi][ni][i] = 0.f;

  const u16* arow[2] = { A + (size_t)(m_base + lr) * K,
                         A + (size_t)(m_base + 16 + lr) * K };
  const u16* wrow[4];
  for (int ni = 0; ni < 4; ++ni)
    wrow[ni] = Wt + (size_t)(n_base + ni * 16 + lr) * K;

  auto load_chunk = [&](int k, v16bf a[2], v16bf wv[4]) {
    for (int mi = 0; mi < 2; ++mi) {
      const u16* p = arow[mi] + k + half * 8;   // elems 0-7: K half*8.., 8-15: +16
      a[mi] = load_frag(p, p + 16);
    }
    for (int ni = 0; ni < 4; ++ni) {
      const u16* p = wrow[ni] + k + half * 8;   // elems 0-7: K half*8.., 8-15: +16
      wv[ni] = load_frag(p, p + 16);
    }
  };
  auto mma_chunk = [&](v16bf a[2], v16bf wv[4]) {
    for (int mi = 0; mi < 2; ++mi)
      for (int ni = 0; ni < 4; ++ni)
        c[mi][ni] = wmma_bf16(a[mi], wv[ni], c[mi][ni]);
  };

  v16bf a0[2], w0[4], a1[2], w1[4];
  load_chunk(0, a0, w0);
  for (int k = 0; k < K; k += 64) {            // K is a multiple of 64
    load_chunk(k + 32, a1, w1);                // prefetch while chunk k computes
    mma_chunk(a0, w0);
    if (k + 64 < K) load_chunk(k + 64, a0, w0);
    mma_chunk(a1, w1);
  }

  for (int mi = 0; mi < 2; ++mi)
    for (int ni = 0; ni < 4; ++ni) {
      int n = n_base + ni * 16 + lr;
      float bv = bias ? bias[n] : 0.f;
      for (int i = 0; i < 8; ++i) {
        int row = m_base + mi * 16 + half * 8 + i;   // C layout: M = 8*half + i
        float vv = c[mi][ni][i] + bv;
        if (act) vv = 0.5f * vv * (1.f + erff(vv * 0.70710678118654752f));
        size_t idx = (size_t)row * N + n;
        if (resid) vv += resid[idx];
        if (outF) outF[idx] = vv;
        if (outB) outB[idx] = f2bf(vv);
      }
    }
}

// ---------------------------------------------------------------------------
// Flash attention, cooperative async K/V staging:
//   One block = 8 waves = 8 consecutive 16-query tiles of the SAME (b,h).
//   Per 32-key chunk the block stages K (32x64) and V^T (64x32) into
//   double-buffered LDS using GLOBAL_LOAD_ASYNC_TO_LDS_B128 (ASYNCcnt) —
//   no VGPR round-trip. Each wave drains its ASYNCcnt before the per-chunk
//   __syncthreads that publishes the buffer; staging of chunk i+1 overlaps
//   compute on chunk i. Cuts global K/V traffic 8x.
//   LDS row strides padded (72 / 40 u16) so 16-lane B-fragment ds_loads hit
//   distinct banks. Online softmax uses 16-lane-segment shuffles (C-fragment
//   rows live per half-wave). P is relaid out C->A via a wave-private LDS
//   slab synchronized with s_wait_dscnt.
// ---------------------------------------------------------------------------
__global__ __launch_bounds__(256) void attn_kernel(
    const u16* __restrict__ qkv, const u16* __restrict__ vt,
    u16* __restrict__ o) {
  const int P = 1024, H = 16;
  const int KS = 72;                            // K tile row stride (u16)
  const int VS = 40;                            // V^T tile row stride (u16)
  __shared__ __align__(16) u16 kbuf[2][32 * KS];
  __shared__ __align__(16) u16 vbuf[2][64 * VS];
  __shared__ __align__(16) u16 pbuf[8][512];    // 16x32 bf16 per wave

  int tid  = threadIdx.x;
  int lane = tid & 31;
  int w    = tid >> 5;
  int lr   = lane & 15, half = lane >> 4;

  int blk = blockIdx.x;                         // 512 blocks
  int qg  = blk & 7;                            // q-group (128 queries)
  int h   = (blk >> 3) & 15;
  int b   = blk >> 7;
  int qt  = qg * 8 + w;                         // this wave's 16-query tile

  const u16* kbase = qkv + (size_t)(b * P) * 3072 + 1024 + h * 64;
  const u16* vbase = vt + (size_t)((b * H + h) * 64) * P;

  // Q fragments (A 16x32): dh 0..31 and 32..63
  const u16* qrow = qkv + (size_t)(b * P + qt * 16 + lr) * 3072 + h * 64;
  v16bf qa[2];
  for (int t = 0; t < 2; ++t) {
    const u16* p = qrow + t * 32 + half * 8;
    qa[t] = load_frag(p, p + 16);
  }

  float m[8], l[8];
  for (int i = 0; i < 8; ++i) { m[i] = -3.0e38f; l[i] = 0.f; }
  v8f oc[4];
  for (int nt = 0; nt < 4; ++nt)
    for (int i = 0; i < 8; ++i) oc[nt][i] = 0.f;

  // Cooperative async staging of one 32-key chunk (all 256 threads, 16B/lane).
  auto stage = [&](int buf, int kk) {
    {   // K: 32 rows x 64 cols, 8 threads per row
      int r = tid >> 3, cb = (tid & 7) * 8;
      async_copy_b128(&kbuf[buf][r * KS + cb],
                      kbase + (size_t)(kk + r) * 3072 + cb);
    }
    {   // V^T: 64 rows x 32 cols, 4 threads per row
      int r = tid >> 2, cb = (tid & 3) * 8;
      async_copy_b128(&vbuf[buf][r * VS + cb],
                      vbase + (size_t)r * P + kk + cb);
    }
  };

  u16* pb = pbuf[w];
  stage(0, 0);
  int buf = 0;

  for (int kk = 0; kk < P; kk += 32) {
    wait_async_zero();                          // drain this wave's async stores
    __syncthreads();                            // staged(buf) visible; buf^1 free
    if (kk + 32 < P) stage(buf ^ 1, kk + 32);   // overlap next-chunk staging

    const u16* kb = kbuf[buf];
    const u16* vb = vbuf[buf];

    // S = Q K^T for 32 keys -> two 16x16 fp32 fragments (chained over DH)
    v8f s[2];
    for (int sf = 0; sf < 2; ++sf) {
      const u16* kp = kb + (size_t)(sf * 16 + lr) * KS;
      v16bf kb0 = load_frag(kp + half * 8,      kp + half * 8 + 16);
      v16bf kb1 = load_frag(kp + 32 + half * 8, kp + 32 + half * 8 + 16);
      v8f acc; for (int i = 0; i < 8; ++i) acc[i] = 0.f;
      acc = wmma_bf16(qa[0], kb0, acc);
      acc = wmma_bf16(qa[1], kb1, acc);
      s[sf] = acc;
    }
    for (int sf = 0; sf < 2; ++sf)
      for (int i = 0; i < 8; ++i) s[sf][i] *= 0.125f;   // DH^-0.5

    // Online softmax: row stats via 16-lane-segment reductions
    float scl[8];
    for (int i = 0; i < 8; ++i) {
      float mx = fmaxf(s[0][i], s[1][i]);
      for (int off = 1; off < 16; off <<= 1)
        mx = fmaxf(mx, __shfl_xor(mx, off, 16));
      float mn = fmaxf(m[i], mx);
      float p0 = __expf(s[0][i] - mn);
      float p1 = __expf(s[1][i] - mn);
      s[0][i] = p0; s[1][i] = p1;
      float rs = p0 + p1;
      for (int off = 1; off < 16; off <<= 1)
        rs += __shfl_xor(rs, off, 16);
      float sc = __expf(m[i] - mn);
      l[i] = l[i] * sc + rs;
      m[i] = mn;
      scl[i] = sc;
    }
    for (int nt = 0; nt < 4; ++nt)
      for (int i = 0; i < 8; ++i) oc[nt][i] *= scl[i];

    // C-layout P -> LDS -> A-layout P (16x32 bf16), wave-private region
    for (int sf = 0; sf < 2; ++sf)
      for (int i = 0; i < 8; ++i)
        pb[(half * 8 + i) * 32 + sf * 16 + lr] = f2bf(s[sf][i]);
    asm volatile("s_wait_dscnt 0" ::: "memory");
    v16bf pa = load_frag(pb + lr * 32 + half * 16,
                         pb + lr * 32 + half * 16 + 8);

    // O += P * V  (V^T tile in LDS: 16 consecutive keys per lane)
    for (int nt = 0; nt < 4; ++nt) {
      const u16* vp = vb + (size_t)(nt * 16 + lr) * VS + half * 16;
      v16bf vf = load_frag(vp, vp + 8);
      oc[nt] = wmma_bf16(pa, vf, oc[nt]);
    }
    buf ^= 1;
  }

  // Normalize and write bf16 output in [B,P,D] layout
  u16* orow = o + (size_t)(b * P + qt * 16) * 1024 + h * 64;
  for (int nt = 0; nt < 4; ++nt)
    for (int i = 0; i < 8; ++i) {
      int r = half * 8 + i;
      orow[(size_t)r * 1024 + nt * 16 + lr] = f2bf(oc[nt][i] / l[i]);
    }
}

// ---------------------------------------------------------------------------
// Host-side orchestration
// ---------------------------------------------------------------------------
extern "C" void kernel_launch(void* const* d_in, const int* in_sizes, int n_in,
                              void* d_out, int out_size, void* d_ws, size_t ws_size,
                              hipStream_t stream) {
  (void)in_sizes; (void)n_in; (void)out_size; (void)ws_size;
  const float* x     = (const float*)d_in[0];
  const float* ln1_g = (const float*)d_in[1];
  const float* ln1_b = (const float*)d_in[2];
  const float* W_qkv = (const float*)d_in[3];
  const float* b_qkv = (const float*)d_in[4];
  const float* W_o   = (const float*)d_in[5];
  const float* b_o   = (const float*)d_in[6];
  const float* ln2_g = (const float*)d_in[7];
  const float* ln2_b = (const float*)d_in[8];
  const float* W1    = (const float*)d_in[9];
  const float* b1    = (const float*)d_in[10];
  const float* W2    = (const float*)d_in[11];
  const float* b2    = (const float*)d_in[12];
  float* out = (float*)d_out;

  const int R = 4096;  // B*P token rows

  char* ws = (char*)d_ws;
  size_t off = 0;
  auto alloc = [&](size_t bytes) -> void* {
    void* p = ws + off;
    off = (off + bytes + 255) & ~(size_t)255;
    return p;
  };
  u16*   xn    = (u16*)  alloc((size_t)R * 1024 * 2);   // LN1 out, bf16
  u16*   wqkvT = (u16*)  alloc((size_t)3072 * 1024 * 2);
  u16*   woT   = (u16*)  alloc((size_t)1024 * 1024 * 2);
  u16*   w1T   = (u16*)  alloc((size_t)4096 * 1024 * 2);
  u16*   w2T   = (u16*)  alloc((size_t)1024 * 4096 * 2);
  u16*   qkv   = (u16*)  alloc((size_t)R * 3072 * 2);   // q|k|v bf16
  u16*   vtb   = (u16*)  alloc((size_t)4 * 16 * 64 * 1024 * 2); // V^T per head
  u16*   ao    = (u16*)  alloc((size_t)R * 1024 * 2);   // attention out, bf16
  float* x1    = (float*)alloc((size_t)R * 1024 * 4);   // attn residual out
  u16*   h2    = (u16*)  alloc((size_t)R * 1024 * 2);   // LN2 out, bf16
  u16*   m1    = (u16*)  alloc((size_t)R * 4096 * 2);   // GELU(h W1 + b1), bf16

  dim3 tb(32, 8);
  // Weight transpose + bf16 convert ([K,N] -> [N,K])
  trans_cvt_kernel<<<dim3(3072/32, 1024/32), tb, 0, stream>>>(W_qkv, wqkvT, 1024, 3072);
  trans_cvt_kernel<<<dim3(1024/32, 1024/32), tb, 0, stream>>>(W_o,   woT,   1024, 1024);
  trans_cvt_kernel<<<dim3(4096/32, 1024/32), tb, 0, stream>>>(W1,    w1T,   1024, 4096);
  trans_cvt_kernel<<<dim3(1024/32, 4096/32), tb, 0, stream>>>(W2,    w2T,   4096, 1024);

  // LN1 -> QKV GEMM
  ln_kernel<<<R, 256, 0, stream>>>(x, ln1_g, ln1_b, xn);
  gemm_kernel<<<dim3(3072/128, R/128), 256, 0, stream>>>(
      xn, wqkvT, b_qkv, nullptr, nullptr, qkv, R, 3072, 1024, 0);

  // Attention
  vtrans_kernel<<<dim3(1024/32, 2, 64), tb, 0, stream>>>(qkv, vtb);
  attn_kernel<<<512, 256, 0, stream>>>(qkv, vtb, ao);

  // O projection + residual (fp32)
  gemm_kernel<<<dim3(1024/128, R/128), 256, 0, stream>>>(
      ao, woT, b_o, x, x1, nullptr, R, 1024, 1024, 0);

  // MLP
  ln_kernel<<<R, 256, 0, stream>>>(x1, ln2_g, ln2_b, h2);
  gemm_kernel<<<dim3(4096/128, R/128), 256, 0, stream>>>(
      h2, w1T, b1, nullptr, nullptr, m1, R, 4096, 1024, 1);
  gemm_kernel<<<dim3(1024/128, R/128), 256, 0, stream>>>(
      m1, w2T, b2, x1, out, nullptr, R, 1024, 4096, 0);
}
